// SingleHeadCausalSelfAttention_87651692576778
// MI455X (gfx1250) — compile-verified, matrix operand order checked by reading, not run on hardware
//
#include <hip/hip_runtime.h>
#include <hip/hip_bf16.h>

// ---------------- CDNA5 WMMA types ----------------
typedef __attribute__((ext_vector_type(16))) _Float16 v16h;
typedef __attribute__((ext_vector_type(8)))  float    v8f;

#define S_LEN   4096
#define D_MODEL 512
#define BATCH   4
#define M_TOT   (BATCH * S_LEN)          // 16384 rows
#define KSTEPS  (D_MODEL / 32)           // 16 k-steps of 32
#define SCALE_QK 0.044194173824159216f   // 1/sqrt(512)

__device__ __forceinline__ v8f v8f_zero() {
    v8f z;
#pragma unroll
    for (int i = 0; i < 8; ++i) z[i] = 0.0f;
    return z;
}

// Load one 16x32 (MxK or NxK) f16 fragment for v_wmma_f32_16x16x32_f16.
// Per ISA layout (wave32): lane l -> half = l>>4, row = l&15.
//   elements 0..7  = K[ 8*half ..  8*half+7]     (one b128 load)
//   elements 8..15 = K[16+8*half .. 16+8*half+7] (one b128 load)
// `tile` points at element [row0][k0] of a row-major matrix, `ld` = row stride (halves).
__device__ __forceinline__ v16h load_frag16(const _Float16* tile, int ld) {
    const int l    = threadIdx.x & 31;
    const int half = l >> 4;
    const int mn   = l & 15;
    const _Float16* p = tile + (size_t)mn * ld + (half << 3);
    v16h f;
    ((uint4*)&f)[0] = *(const uint4*)(p);
    ((uint4*)&f)[1] = *(const uint4*)(p + 16);
    return f;
}

// ---------------- stage 0: fp32 -> f16 ----------------
__global__ void cvt_f32_to_f16(const float* __restrict__ src,
                               _Float16* __restrict__ dst, int n) {
    int i = blockIdx.x * blockDim.x + threadIdx.x;
    int stride = gridDim.x * blockDim.x;
    for (; i < n; i += stride) dst[i] = (_Float16)src[i];
}

// ---------------- stage 1: fused QKV projection ----------------
// q/k/v[m][e] = sum_d x[m][d] * W[e][d].  One wave -> 16(M) x 64(N) tile.
// Fully unrolled, parity-ping-pong pipelined k-loop: loads for step ks+1 are
// issued before the 4 WMMAs of step ks; unrolling makes the ping-pong buffers
// distinct registers (no rotation movs, no WMMA->VALU hazard nops).
// V is stored transposed: vT[b][dim][s].
__global__ __launch_bounds__(256) void qkv_gemm(
    const _Float16* __restrict__ xh,
    const _Float16* __restrict__ wqh,
    const _Float16* __restrict__ wkh,
    const _Float16* __restrict__ wvh,
    _Float16* __restrict__ qh,
    _Float16* __restrict__ kh,
    _Float16* __restrict__ vT) {

    const int wave = threadIdx.x >> 5;
    const int lane = threadIdx.x & 31;
    const int half = lane >> 4;
    const int nn   = lane & 15;

    const int m0 = (blockIdx.x * 8 + wave) * 16;
    const int n0 = blockIdx.y * 64;
    const int z  = blockIdx.z;
    const _Float16* W = (z == 0) ? wqh : (z == 1) ? wkh : wvh;

    const _Float16* Abase = xh + (size_t)m0 * D_MODEL;

    v8f acc[4];
#pragma unroll
    for (int nt = 0; nt < 4; ++nt) acc[nt] = v8f_zero();

    v16h abuf[2];
    v16h bbuf[2][4];
    abuf[0] = load_frag16(Abase, D_MODEL);
#pragma unroll
    for (int nt = 0; nt < 4; ++nt)
        bbuf[0][nt] = load_frag16(W + (size_t)(n0 + nt * 16) * D_MODEL, D_MODEL);

#pragma unroll
    for (int ks = 0; ks < KSTEPS; ++ks) {
        const int cur = ks & 1;
        const int nxt = cur ^ 1;
        if (ks + 1 < KSTEPS) {
            const int k1 = (ks + 1) * 32;
            abuf[nxt] = load_frag16(Abase + k1, D_MODEL);
#pragma unroll
            for (int nt = 0; nt < 4; ++nt)
                bbuf[nxt][nt] =
                    load_frag16(W + (size_t)(n0 + nt * 16) * D_MODEL + k1, D_MODEL);
        }
#pragma unroll
        for (int nt = 0; nt < 4; ++nt)
            acc[nt] = __builtin_amdgcn_wmma_f32_16x16x32_f16(
                false, abuf[cur], false, bbuf[cur][nt], (short)0, acc[nt], false, false);
    }

#pragma unroll
    for (int nt = 0; nt < 4; ++nt) {
#pragma unroll
        for (int r = 0; r < 8; ++r) {
            const int row = m0 + r + 8 * half;           // global m
            const int col = n0 + nt * 16 + nn;           // output feature
            const _Float16 h = (_Float16)acc[nt][r];
            if (z == 0) {
                qh[(size_t)row * D_MODEL + col] = h;
            } else if (z == 1) {
                kh[(size_t)row * D_MODEL + col] = h;
            } else {
                const int b = row >> 12;                 // row / 4096
                const int s = row & (S_LEN - 1);
                vT[((size_t)b * D_MODEL + col) * S_LEN + s] = h;
            }
        }
    }
}

// ---------------- stage 2: flash attention (causal, online softmax) ----------------
// Workgroup (256 thr, 8 waves) owns 16 queries. Wave w owns output dims [64w, 64w+64).
__global__ __launch_bounds__(256) void flash_attn(
    const _Float16* __restrict__ qh,
    const _Float16* __restrict__ kh,
    const _Float16* __restrict__ vT,
    _Float16* __restrict__ yh) {

    const int tid  = threadIdx.x;
    const int wave = tid >> 5;
    const int lane = tid & 31;
    const int half = lane >> 4;
    const int nn   = lane & 15;

    const int qt = blockIdx.x;
    const int b  = qt >> 8;                 // / (S/16)
    const int q0 = (qt & 255) * 16;

    __shared__ __align__(16) float    sS[16 * 64];   // partial scores (f32)
    __shared__ __align__(16) _Float16 sP[16 * 64];   // probabilities (f16)
    __shared__ float sMax[16], sSum[16], sAlpha[16];
    __shared__ float sPM[64];                        // per-quarter row maxima

    // Q fragments for this wave's 64-wide D slice (kept in registers).
    const _Float16* qbase = qh + ((size_t)(b * S_LEN + q0)) * D_MODEL + wave * 64;
    v16h qf[2];
#pragma unroll
    for (int kk = 0; kk < 2; ++kk) qf[kk] = load_frag16(qbase + kk * 32, D_MODEL);

    v8f o[4];
#pragma unroll
    for (int nt = 0; nt < 4; ++nt) o[nt] = v8f_zero();

    if (tid < 16) { sMax[tid] = -1e30f; sSum[tid] = 0.0f; }
    __syncthreads();

    for (int kb = 0; kb < q0 + 16; kb += 64) {
        // prefetch next key block's K rows into cache (global_prefetch_b8)
        if (kb + 64 < q0 + 16) {
            const _Float16* pf =
                kh + ((size_t)(b * S_LEN + kb + 64 + (lane & 31))) * D_MODEL + wave * 64;
            __builtin_prefetch(pf, 0, 3);
            __builtin_prefetch(pf + (size_t)32 * D_MODEL, 0, 3);
        }

        // zero the score tile
        for (int i = tid; i < 16 * 64; i += 256) sS[i] = 0.0f;
        __syncthreads();

        // partial scores over this wave's D slice: S += Q_slice @ K_slice^T
#pragma unroll
        for (int nt = 0; nt < 4; ++nt) {
            v8f c = v8f_zero();
#pragma unroll
            for (int kk = 0; kk < 2; ++kk) {
                v16h kf = load_frag16(
                    kh + ((size_t)(b * S_LEN + kb + nt * 16)) * D_MODEL + wave * 64 + kk * 32,
                    D_MODEL);
                c = __builtin_amdgcn_wmma_f32_16x16x32_f16(
                    false, qf[kk], false, kf, (short)0, c, false, false);
            }
#pragma unroll
            for (int r = 0; r < 8; ++r)
                atomicAdd(&sS[(r + 8 * half) * 64 + nt * 16 + nn], c[r]);
        }
        __syncthreads();

        // ---- parallel online softmax ----
        // (a) 64 threads: partial masked row maxima (row = t>>2, 16 cols each)
        if (tid < 64) {
            const int m   = tid >> 2;
            const int j0  = (tid & 3) * 16;
            const int mg  = q0 + m;
            float bm = -1e30f;
            for (int j = 0; j < 16; ++j) {
                const int jg = kb + j0 + j;
                if (jg <= mg) bm = fmaxf(bm, sS[m * 64 + j0 + j] * SCALE_QK);
            }
            sPM[tid] = bm;
        }
        __syncthreads();
        // (b) 16 threads: merge stats, fold rescale into running sum
        if (tid < 16) {
            const int m = tid;
            float bm = fmaxf(fmaxf(sPM[m * 4 + 0], sPM[m * 4 + 1]),
                             fmaxf(sPM[m * 4 + 2], sPM[m * 4 + 3]));
            const float nm    = fmaxf(sMax[m], bm);
            const float alpha = __expf(sMax[m] - nm);
            sMax[m]   = nm;
            sAlpha[m] = alpha;
            sSum[m]  *= alpha;
        }
        __syncthreads();
        // (c) 256 threads: masked exp (4 elements each) + partial row sums
        {
            const int m  = tid >> 4;
            const int j0 = (tid & 15) * 4;
            const int mg = q0 + m;
            const float nm = sMax[m];
            float psum = 0.0f;
#pragma unroll
            for (int jj = 0; jj < 4; ++jj) {
                const int jg = kb + j0 + jj;
                float p = 0.0f;
                if (jg <= mg) p = __expf(sS[m * 64 + j0 + jj] * SCALE_QK - nm);
                sP[m * 64 + j0 + jj] = (_Float16)p;
                psum += p;
            }
            atomicAdd(&sSum[m], psum);
        }
        __syncthreads();

        // rescale running O, then O += P @ V_slice
#pragma unroll
        for (int r = 0; r < 8; ++r) {
            const float al = sAlpha[r + 8 * half];
#pragma unroll
            for (int nt = 0; nt < 4; ++nt) o[nt][r] *= al;
        }
#pragma unroll
        for (int kk = 0; kk < 2; ++kk) {
            v16h pfr = load_frag16(&sP[kk * 32], 64);   // P: 16 queries x 64 keys (LDS)
#pragma unroll
            for (int nt = 0; nt < 4; ++nt) {
                v16h vf = load_frag16(
                    vT + ((size_t)b * D_MODEL + wave * 64 + nt * 16) * S_LEN + kb + kk * 32,
                    S_LEN);
                o[nt] = __builtin_amdgcn_wmma_f32_16x16x32_f16(
                    false, pfr, false, vf, (short)0, o[nt], false, false);
            }
        }
        __syncthreads();
    }

    // normalize and write y (f16)
#pragma unroll
    for (int r = 0; r < 8; ++r) {
        const float inv = 1.0f / sSum[r + 8 * half];
        const int row = q0 + r + 8 * half;
#pragma unroll
        for (int nt = 0; nt < 4; ++nt) {
            const int col = wave * 64 + nt * 16 + nn;
            yh[((size_t)(b * S_LEN + row)) * D_MODEL + col] = (_Float16)(o[nt][r] * inv);
        }
    }
}

// ---------------- stage 3: output projection + bias (fp32 out) ----------------
__global__ __launch_bounds__(256) void out_gemm(
    const _Float16* __restrict__ yh,
    const _Float16* __restrict__ woh,
    const float* __restrict__ bo,
    float* __restrict__ out) {

    const int wave = threadIdx.x >> 5;
    const int lane = threadIdx.x & 31;
    const int half = lane >> 4;
    const int nn   = lane & 15;

    const int m0 = (blockIdx.x * 8 + wave) * 16;
    const int n0 = blockIdx.y * 64;

    const _Float16* Abase = yh + (size_t)m0 * D_MODEL;

    v8f acc[4];
#pragma unroll
    for (int nt = 0; nt < 4; ++nt) acc[nt] = v8f_zero();

    v16h abuf[2];
    v16h bbuf[2][4];
    abuf[0] = load_frag16(Abase, D_MODEL);
#pragma unroll
    for (int nt = 0; nt < 4; ++nt)
        bbuf[0][nt] = load_frag16(woh + (size_t)(n0 + nt * 16) * D_MODEL, D_MODEL);

#pragma unroll
    for (int ks = 0; ks < KSTEPS; ++ks) {
        const int cur = ks & 1;
        const int nxt = cur ^ 1;
        if (ks + 1 < KSTEPS) {
            const int k1 = (ks + 1) * 32;
            abuf[nxt] = load_frag16(Abase + k1, D_MODEL);
#pragma unroll
            for (int nt = 0; nt < 4; ++nt)
                bbuf[nxt][nt] =
                    load_frag16(woh + (size_t)(n0 + nt * 16) * D_MODEL + k1, D_MODEL);
        }
#pragma unroll
        for (int nt = 0; nt < 4; ++nt)
            acc[nt] = __builtin_amdgcn_wmma_f32_16x16x32_f16(
                false, abuf[cur], false, bbuf[cur][nt], (short)0, acc[nt], false, false);
    }

#pragma unroll
    for (int nt = 0; nt < 4; ++nt) {
        const int col = n0 + nt * 16 + nn;
        const float bias = bo[col];
#pragma unroll
        for (int r = 0; r < 8; ++r) {
            const int row = m0 + r + 8 * half;
            out[(size_t)row * D_MODEL + col] = acc[nt][r] + bias;
        }
    }
}

// ---------------- launcher ----------------
extern "C" void kernel_launch(void* const* d_in, const int* in_sizes, int n_in,
                              void* d_out, int out_size, void* d_ws, size_t ws_size,
                              hipStream_t stream) {
    (void)in_sizes; (void)n_in; (void)out_size; (void)ws_size;

    const float* x  = (const float*)d_in[0];
    const float* Wq = (const float*)d_in[1];
    const float* Wk = (const float*)d_in[2];
    const float* Wv = (const float*)d_in[3];
    const float* Wo = (const float*)d_in[4];
    const float* bo = (const float*)d_in[5];
    float* out = (float*)d_out;

    const size_t NX = (size_t)M_TOT * D_MODEL;     // 8,388,608
    const size_t NW = (size_t)D_MODEL * D_MODEL;   // 262,144

    _Float16* xh  = (_Float16*)d_ws;
    _Float16* qh  = xh  + NX;
    _Float16* kh  = qh  + NX;
    _Float16* vT  = kh  + NX;
    _Float16* yh  = vT  + NX;
    _Float16* wqh = yh  + NX;
    _Float16* wkh = wqh + NW;
    _Float16* wvh = wkh + NW;
    _Float16* woh = wvh + NW;
    // total ws use: (5*NX + 4*NW) * 2 bytes ~= 86 MB

    // stage 0: fp32 -> f16
    cvt_f32_to_f16<<<(int)((NX + 255) / 256), 256, 0, stream>>>(x,  xh,  (int)NX);
    cvt_f32_to_f16<<<(int)((NW + 255) / 256), 256, 0, stream>>>(Wq, wqh, (int)NW);
    cvt_f32_to_f16<<<(int)((NW + 255) / 256), 256, 0, stream>>>(Wk, wkh, (int)NW);
    cvt_f32_to_f16<<<(int)((NW + 255) / 256), 256, 0, stream>>>(Wv, wvh, (int)NW);
    cvt_f32_to_f16<<<(int)((NW + 255) / 256), 256, 0, stream>>>(Wo, woh, (int)NW);

    // stage 1: QKV projections (grid: 128 m-tile groups x 8 n-chunks x {q,k,v})
    qkv_gemm<<<dim3(M_TOT / 16 / 8, D_MODEL / 64, 3), 256, 0, stream>>>(
        xh, wqh, wkh, wvh, qh, kh, vT);

    // stage 2: flash attention, one block per 16-query tile
    flash_attn<<<dim3(BATCH * (S_LEN / 16)), 256, 0, stream>>>(qh, kh, vT, yh);

    // stage 3: output projection + bias
    out_gemm<<<dim3(M_TOT / 16 / 8, D_MODEL / 64), 256, 0, stream>>>(yh, woh, bo, out);
}
